// TwoWordBilinearLabelProbe_4148938408320
// MI455X (gfx1250) — matrix operand order repaired and checked by reading, not compiled, not run on hardware
//
#include <hip/hip_runtime.h>

// ---------------------------------------------------------------------------
// Types matching the gfx1250 WMMA builtin signatures (probe-confirmed).
// ---------------------------------------------------------------------------
typedef __attribute__((ext_vector_type(16))) __bf16      v16bf;
typedef __attribute__((ext_vector_type(8)))  float        v8f;
typedef __attribute__((ext_vector_type(4)))  unsigned int v4u;
typedef __attribute__((ext_vector_type(2)))  unsigned int v2u;
typedef __attribute__((ext_vector_type(4)))  float        v4f;

struct Q32 { v4u x, y; };   // 32 bytes == one v16bf fragment

#define B_DIM 8
#define S_DIM 1024
#define D_DIM 1024
#define R_DIM 256
#define KTILE 32
#define LDS_STRIDE 40   // 32 ushorts payload + 8 pad: 80B rows, 16B-aligned frags

// sched_group_barrier masks (LLVM AMDGPU)
#define SGB_MFMA      0x008u   // MFMA/WMMA class
#define SGB_VMEM_READ 0x020u
#define SGB_DS_READ   0x100u

static __device__ __forceinline__ unsigned short cvt_bf16(float f) {
  // round-to-nearest-even f32 -> bf16
  unsigned int u = __builtin_bit_cast(unsigned int, f);
  u += 0x7FFFu + ((u >> 16) & 1u);
  return (unsigned short)(u >> 16);
}

static __device__ __forceinline__ v2u pack4_bf16(v4f f) {
  v2u p;
  p.x = ((unsigned int)cvt_bf16(f.y) << 16) | (unsigned int)cvt_bf16(f.x);
  p.y = ((unsigned int)cvt_bf16(f.w) << 16) | (unsigned int)cvt_bf16(f.z);
  return p;
}

// Build a fragment from two 16B-aligned halves (LDS or global/L2 resident).
static __device__ __forceinline__ v16bf load_frag(const unsigned short* p0,
                                                  const unsigned short* p1) {
  Q32 q;
  q.x = *(const v4u*)p0;
  q.y = *(const v4u*)p1;
  return __builtin_bit_cast(v16bf, q);
}

// ---------------------------------------------------------------------------
// Stage 0: one-time f32 -> bf16 conversion of the projections into workspace.
//   pLT[n][k] = bf16(proj_L[k][n])   (transposed -> K-contiguous B fragments)
//   pRT[n][k] = bf16(proj_R[n][k])   (already n-major)
// 1 MB total; stays resident in L2/WGP$ for all of stage 1.
// ---------------------------------------------------------------------------
__global__ __launch_bounds__(256) void stage0_cvt(
    const float* __restrict__ projL, const float* __restrict__ projR,
    unsigned short* __restrict__ pLT, unsigned short* __restrict__ pRT) {
  const int t  = threadIdx.x;        // n = t in [0,256)
  const int k0 = blockIdx.x * 32;    // 32 blocks cover D=1024
#pragma unroll
  for (int k = 0; k < 32; k += 4) {  // transpose: coalesced reads across n
    v4f f;
    f.x = projL[(size_t)(k0 + k + 0) * R_DIM + t];
    f.y = projL[(size_t)(k0 + k + 1) * R_DIM + t];
    f.z = projL[(size_t)(k0 + k + 2) * R_DIM + t];
    f.w = projL[(size_t)(k0 + k + 3) * R_DIM + t];
    *(v2u*)&pLT[(size_t)t * D_DIM + k0 + k] = pack4_bf16(f);
  }
#pragma unroll
  for (int c4 = 0; c4 < 8; ++c4) {
    v4f f = *(const v4f*)&projR[(size_t)t * D_DIM + k0 + c4 * 4];
    *(v2u*)&pRT[(size_t)t * D_DIM + k0 + c4 * 4] = pack4_bf16(f);
  }
}

// ---------------------------------------------------------------------------
// Stage 1: left = X @ L, right = X @ R^T   (X is 8192x1024 f32, read ONCE).
// 512 threads (16 waves), 8 accumulators per wave (sel x col-half x row-tile)
// for high occupancy on the HBM stream. LDS holds only the shared bf16 X tile
// (5 KB); B fragments stream from the L2-resident bf16 projections.
// sched_group_barrier pipelines the compute region: A-frag + 2 B tiles in
// flight before the first WMMA, then 1 WMMA per 2 loads.
// ---------------------------------------------------------------------------
__global__ __launch_bounds__(512) void stage1_proj(
    const float* __restrict__ batch, const unsigned short* __restrict__ pLT,
    const unsigned short* __restrict__ pRT, unsigned short* __restrict__ wsL,
    unsigned short* __restrict__ wsR) {
  __shared__ __align__(16) unsigned short ldsA[64][LDS_STRIDE];

  const int t    = threadIdx.x;
  const int lane = t & 31;
  const int wave = t >> 5;            // 0..15
  const int sel  = wave >> 3;         // 0 -> left output, 1 -> right output
  const int ch   = (wave >> 2) & 1;   // column half: cols [ch*128, ch*128+128)
  const int rt   = wave & 3;          // 16-row tile within the 64-row stripe
  const int m0   = blockIdx.x * 64;   // global row (b*1024+s flattened)

  const v8f vzero = {0.f, 0.f, 0.f, 0.f, 0.f, 0.f, 0.f, 0.f};
  v8f acc[8];
#pragma unroll
  for (int i = 0; i < 8; ++i) acc[i] = vzero;

  const int rowA  = rt * 16 + (lane & 15);
  const int baseA = (lane >= 16) ? 8 : 0;    // A-frag K-half select (ISA layout)
  const int kbB   = (lane >= 16) ? 8 : 0;    // B-frag K-half select
  const int colB  = lane & 15;
  const unsigned short* projB = sel ? pRT : pLT;   // [n][k] bf16, K-contiguous
  const int nBase = ch * 128 + colB;

  for (int k0 = 0; k0 < D_DIM; k0 += KTILE) {
    // --- X tile: 64x32 f32 -> bf16 into LDS (one float4 per thread) ---
    {
      int row = t >> 3;
      int c4  = (t & 7) * 4;
      v4f f = *(const v4f*)&batch[(size_t)(m0 + row) * D_DIM + k0 + c4];
      *(v2u*)&ldsA[row][c4] = pack4_bf16(f);
      if (k0 + KTILE < D_DIM) {  // global_prefetch_b8 on the next HBM slab
        __builtin_prefetch(&batch[(size_t)(m0 + row) * D_DIM + k0 + KTILE], 0, 0);
      }
    }
    __syncthreads();

    v16bf a = load_frag(&ldsA[rowA][baseA], &ldsA[rowA][baseA + 16]);

    // B fragments streamed from L2, pipelined one tile ahead of the WMMA.
    const unsigned short* bp0 = &projB[(size_t)nBase * D_DIM + k0 + kbB];
    v16bf bcur = load_frag(bp0, bp0 + 16);
#pragma unroll
    for (int ct = 0; ct < 8; ++ct) {
      v16bf bnext = bcur;
      if (ct + 1 < 8) {
        const unsigned short* bp =
            &projB[(size_t)(nBase + (ct + 1) * 16) * D_DIM + k0 + kbB];
        bnext = load_frag(bp, bp + 16);
      }
      acc[ct] = __builtin_amdgcn_wmma_f32_16x16x32_bf16(
          false, a, false, bcur, (short)0, acc[ct], false, false);
      bcur = bnext;
    }

    // Compute-region schedule (between the two barriers):
    // 2 DS reads (A frag), 16 VMEM reads (B), 8 WMMAs.
    // [DS x2][V x4] ([M x1][V x2]) x6 [M x2]
    __builtin_amdgcn_sched_group_barrier(SGB_DS_READ, 2, 0);
    __builtin_amdgcn_sched_group_barrier(SGB_VMEM_READ, 4, 0);
#pragma unroll
    for (int g = 0; g < 6; ++g) {
      __builtin_amdgcn_sched_group_barrier(SGB_MFMA, 1, 0);
      __builtin_amdgcn_sched_group_barrier(SGB_VMEM_READ, 2, 0);
    }
    __builtin_amdgcn_sched_group_barrier(SGB_MFMA, 2, 0);

    __syncthreads();
  }

  // Store accumulators as bf16 (C/D layout: lane = col, 8 rows per half-wave).
  unsigned short* outp = sel ? wsR : wsL;
  const int rb = (lane >= 16) ? 8 : 0;
#pragma unroll
  for (int ct = 0; ct < 8; ++ct) {
#pragma unroll
    for (int v = 0; v < 8; ++v) {
      int m = m0 + rt * 16 + rb + v;
      outp[(size_t)m * R_DIM + nBase + ct * 16] = cvt_bf16(acc[ct][v]);
    }
  }
}

// ---------------------------------------------------------------------------
// Stage 2: logits[b] = left[b] @ right[b]^T + bias  (1024x256 @ 256x1024, x8)
// Entire working set (8 MB bf16) is L2-resident; all fragments are
// two K-contiguous 16B runs (k 8h+s, 16+8h+s) -> stream straight from L2 with global_load_b128
// pairs. No LDS, no barriers. A fragment double-buffered across k-iterations;
// sched_group_barrier keeps 2 B tiles in flight ahead of every WMMA.
// ---------------------------------------------------------------------------
__global__ __launch_bounds__(256) void stage2_gram(
    const unsigned short* __restrict__ wsL, const unsigned short* __restrict__ wsR,
    const float* __restrict__ bias, float* __restrict__ out) {
  const int t    = threadIdx.x;
  const int lane = t & 31;
  const int wave = t >> 5;
  const int b    = blockIdx.z;
  const int i0   = blockIdx.y * 128;
  const int j0   = blockIdx.x * 128;

  const v8f vzero = {0.f, 0.f, 0.f, 0.f, 0.f, 0.f, 0.f, 0.f};
  v8f acc[8];
#pragma unroll
  for (int i = 0; i < 8; ++i) acc[i] = vzero;

  const int baseA = (lane >= 16) ? 8 : 0;
  const int kbB   = (lane >= 16) ? 8 : 0;
  const int colB  = lane & 15;

  const unsigned short* aRow =
      &wsL[((size_t)b * S_DIM + i0 + wave * 16 + (lane & 15)) * R_DIM];
  const unsigned short* bRow = &wsR[((size_t)b * S_DIM + j0 + colB) * R_DIM];

  v16bf a_cur = load_frag(aRow + baseA, aRow + baseA + 16);

#pragma unroll 1
  for (int r0 = 0; r0 < R_DIM; r0 += KTILE) {
    // A fragment for the NEXT k-iteration issues inside this region.
    v16bf a_next = a_cur;
    if (r0 + KTILE < R_DIM) {
      a_next = load_frag(aRow + r0 + KTILE + baseA, aRow + r0 + KTILE + baseA + 16);
    }

    const unsigned short* bp0 = bRow + r0 + kbB;
    v16bf bcur = load_frag(bp0, bp0 + 16);
#pragma unroll
    for (int ct = 0; ct < 8; ++ct) {
      v16bf bnext = bcur;
      if (ct + 1 < 8) {
        const unsigned short* bp = bRow + (size_t)(ct + 1) * 16 * R_DIM + r0 + kbB;
        bnext = load_frag(bp, bp + 16);
      }
      acc[ct] = __builtin_amdgcn_wmma_f32_16x16x32_bf16(
          false, a_cur, false, bcur, (short)0, acc[ct], false, false);
      bcur = bnext;
    }
    a_cur = a_next;

    // Per-iteration schedule: 18 VMEM reads (2 next-A + 16 B), 8 WMMAs.
    // [V x6] ([M x1][V x2]) x6 [M x2]
    __builtin_amdgcn_sched_group_barrier(SGB_VMEM_READ, 6, 0);
#pragma unroll
    for (int g = 0; g < 6; ++g) {
      __builtin_amdgcn_sched_group_barrier(SGB_MFMA, 1, 0);
      __builtin_amdgcn_sched_group_barrier(SGB_VMEM_READ, 2, 0);
    }
    __builtin_amdgcn_sched_group_barrier(SGB_MFMA, 2, 0);
  }

  const float bv = bias[0];
  const int rb = (lane >= 16) ? 8 : 0;
#pragma unroll
  for (int ct = 0; ct < 8; ++ct) {
#pragma unroll
    for (int v = 0; v < 8; ++v) {
      size_t oi = ((size_t)b << 20) +
                  (size_t)(i0 + wave * 16 + rb + v) * S_DIM +
                  (size_t)(j0 + ct * 16 + colB);
      out[oi] = acc[ct][v] + bv;
    }
  }
}

// ---------------------------------------------------------------------------
extern "C" void kernel_launch(void* const* d_in, const int* in_sizes, int n_in,
                              void* d_out, int out_size, void* d_ws, size_t ws_size,
                              hipStream_t stream) {
  const float* batch = (const float*)d_in[0];   // (8,1024,1024) f32
  const float* projL = (const float*)d_in[1];   // (1024,256)    f32
  const float* projR = (const float*)d_in[2];   // (256,1024)    f32
  const float* bias  = (const float*)d_in[3];   // (1,)          f32
  float* out = (float*)d_out;                   // (8,1024,1024) f32

  // Workspace layout (bf16): left 4MB | right 4MB | projLT 0.5MB | projRT 0.5MB
  unsigned short* wsL = (unsigned short*)d_ws;
  unsigned short* wsR = wsL + (size_t)B_DIM * S_DIM * R_DIM;
  unsigned short* pLT = wsR + (size_t)B_DIM * S_DIM * R_DIM;
  unsigned short* pRT = pLT + (size_t)R_DIM * D_DIM;

  hipLaunchKernelGGL(stage0_cvt, dim3(D_DIM / 32), dim3(256), 0, stream,
                     projL, projR, pLT, pRT);
  hipLaunchKernelGGL(stage1_proj, dim3((B_DIM * S_DIM) / 64), dim3(512), 0, stream,
                     batch, pLT, pRT, wsL, wsR);
  hipLaunchKernelGGL(stage2_gram, dim3(S_DIM / 128, S_DIM / 128, B_DIM), dim3(256),
                     0, stream, wsL, wsR, bias, out);
}
